// DecoderLayer_86036784873883
// MI455X (gfx1250) — matrix operand order repaired and checked by reading, not compiled, not run on hardware
//
#include <hip/hip_runtime.h>

// ---------------- CDNA5 feature probes ----------------
#if defined(__AMDGCN__) && __has_builtin(__builtin_amdgcn_global_load_async_to_lds_b128)
#define HAVE_ASYNC_LDS 1
#else
#define HAVE_ASYNC_LDS 0
#endif

#if defined(__AMDGCN__) && __has_builtin(__builtin_amdgcn_tensor_load_to_lds)
#define HAVE_TDM 1
#else
#define HAVE_TDM 0
#endif

#define USE_TDM (HAVE_TDM && HAVE_ASYNC_LDS)

__device__ __forceinline__ void wait_async0() {
#if defined(__AMDGCN__)
#if __has_builtin(__builtin_amdgcn_s_wait_asynccnt)
  __builtin_amdgcn_s_wait_asynccnt(0);
#else
  asm volatile("s_wait_asynccnt 0x0" ::: "memory");
#endif
#endif
}

__device__ __forceinline__ void wait_tensor0() {
#if defined(__AMDGCN__)
#if __has_builtin(__builtin_amdgcn_s_wait_tensorcnt)
  __builtin_amdgcn_s_wait_tensorcnt(0);
#else
  asm volatile("s_wait_tensorcnt 0x0" ::: "memory");
#endif
#endif
}

#if HAVE_ASYNC_LDS
// async copy of 16B from global to LDS (per-lane), tracked by ASYNCcnt
typedef int gv4i __attribute__((vector_size(16)));
typedef __attribute__((address_space(1))) gv4i* gv4i_gp;
typedef __attribute__((address_space(3))) gv4i* gv4i_lp;
__device__ __forceinline__ void async_cp16(const unsigned short* g, unsigned short* l) {
  __builtin_amdgcn_global_load_async_to_lds_b128((gv4i_gp)g, (gv4i_lp)l, 0, 0);
}
#endif

#if USE_TDM
// Tensor Data Mover: DMA a 128-row x 32-elem bf16 tile of a row-major [N][ldb]
// tensor into LDS with 16B padding after every 64B row (=> 80B LDS row stride).
typedef unsigned int u32x4 __attribute__((vector_size(16)));
typedef int i32x8 __attribute__((vector_size(32)));
typedef int i32x4 __attribute__((vector_size(16)));

__device__ __forceinline__ unsigned lds_addr_of(const void* p) {
  return (unsigned)(unsigned long long)(__attribute__((address_space(3))) const char*)p;
}

__device__ __forceinline__ void tdm_load_B(const unsigned short* gsrc, const unsigned short* ldst,
                                           int K, int Nrows, int ldb) {
  unsigned long long ga = (unsigned long long)gsrc;
  unsigned la = lds_addr_of(ldst);
  // group0: count=1 | lds_addr | global_addr[56:0] | type=2
  u32x4 g0 = { 1u, la, (unsigned)(ga & 0xffffffffu),
               (unsigned)((ga >> 32) & 0x01ffffffu) | (2u << 30) };
  // group1: data_size=2B(1), pad_enable, pad_interval=16dw(3), pad_amount=4dw(3),
  // tensor_dim0=K @bit48, tensor_dim1=Nrows @bit80, tile_dim0=32 @bit112,
  // tile_dim1=128 @bit128, tensor_dim0_stride=ldb @bit160
  int w0 = (1 << 16) | (1 << 20) | (3 << 22) | (3 << 25);
  i32x8 g1 = { w0,
               (int)(((unsigned)K & 0xffffu) << 16),
               (int)(((unsigned)K >> 16) | (((unsigned)Nrows & 0xffffu) << 16)),
               (int)(((unsigned)Nrows >> 16) | (32u << 16)),
               128, ldb, 0, 0 };
  i32x4 z4 = { 0, 0, 0, 0 };
  i32x8 z8 = { 0, 0, 0, 0, 0, 0, 0, 0 };
  __builtin_amdgcn_tensor_load_to_lds(g0, g1, z4, z4, z8, 0);
}
#endif

// ---------------- types ----------------
typedef __attribute__((ext_vector_type(16))) __bf16 v16bf;
typedef __attribute__((ext_vector_type(8)))  float  v8f;

union FragAB { v16bf v; uint4 u[2]; };

__device__ __forceinline__ unsigned short f2bf(float f) {
  unsigned int u = __float_as_uint(f);
  u += 0x7fffu + ((u >> 16) & 1u);           // round-to-nearest-even
  return (unsigned short)(u >> 16);
}
__device__ __forceinline__ float bf2f(unsigned short h) {
  return __uint_as_float(((unsigned int)h) << 16);
}

#define NEGC (-10000.0f)

// ---------------- f32 -> bf16 conversion ----------------
__global__ __launch_bounds__(256)
void k_cvt(const float* __restrict__ src, unsigned short* __restrict__ dst, int n4) {
  int i = blockIdx.x * 256 + threadIdx.x;
  if (i < n4) {
    float4 v = ((const float4*)src)[i];
    ushort4 o;
    o.x = f2bf(v.x); o.y = f2bf(v.y); o.z = f2bf(v.z); o.w = f2bf(v.w);
    ((ushort4*)dst)[i] = o;
  }
}

// ---------------- rmsnorm (row of 2048) -> bf16 ----------------
__global__ __launch_bounds__(256)
void k_rmsnorm(const float* __restrict__ x, const float* __restrict__ w,
               unsigned short* __restrict__ out) {
  __shared__ float red[8];
  int row = blockIdx.x, t = threadIdx.x;
  const float4* xr = (const float4*)(x + (size_t)row * 2048);
  float4 a = xr[t], b = xr[t + 256];
  float ss = a.x*a.x + a.y*a.y + a.z*a.z + a.w*a.w
           + b.x*b.x + b.y*b.y + b.z*b.z + b.w*b.w;
  #pragma unroll
  for (int m = 16; m; m >>= 1) ss += __shfl_xor(ss, m, 32);
  if ((t & 31) == 0) red[t >> 5] = ss;
  __syncthreads();
  float tot = 0.f;
  #pragma unroll
  for (int i = 0; i < 8; ++i) tot += red[i];
  float rms = rsqrtf(tot * (1.f / 2048.f) + 1.1920928955078125e-07f);
  const float4* wr = (const float4*)w;
  float4 wa = wr[t], wb = wr[t + 256];
  ushort4* o = (ushort4*)(out + (size_t)row * 2048);
  ushort4 oa, ob;
  oa.x = f2bf(a.x * rms * wa.x); oa.y = f2bf(a.y * rms * wa.y);
  oa.z = f2bf(a.z * rms * wa.z); oa.w = f2bf(a.w * rms * wa.w);
  ob.x = f2bf(b.x * rms * wb.x); ob.y = f2bf(b.y * rms * wb.y);
  ob.z = f2bf(b.z * rms * wb.z); ob.w = f2bf(b.w * rms * wb.w);
  o[t] = oa; o[t + 256] = ob;
}

// ---------------- generic bf16 NT GEMM: C[M,N] = A[M,K] * B[N,K]^T ----------------
// 128x128 block tile, BK=32, 8 waves, each wave 32x64 (2x4 WMMA frags).
// A tiles: per-lane GLOBAL_LOAD_ASYNC_TO_LDS_B128 (supports row gather).
// B tiles: TENSOR_LOAD_TO_LDS DMA (TDM, TENSORcnt) when available.
struct GemmArgs {
  const unsigned short* A; int lda;
  const unsigned short* Bw; int ldb;
  int M, N, K;
  const int* dynM;          // optional device row count
  const int* rowIdx;        // optional A-row gather
  // EPI 0: qkv split
  unsigned short* q; unsigned short* kk; unsigned short* v;
  // EPI 1: +resid -> f32 x1 and d_out
  float* x1; float* out2; const float* resid;
  // EPI 2: silu -> bf16
  unsigned short* hx;
  // EPI 3: weighted atomic scatter
  const float* slotW; const int* slotTok; float* atom;
};

#define BM 128
#define BN 128
#define BK 32
#define LDS_S 40    // padded row stride (bf16 elems)

template<int EPI>
__global__ __launch_bounds__(256)
void k_gemm(GemmArgs g) {
  __shared__ unsigned short As[2][BM * LDS_S];
  __shared__ unsigned short Bs[2][BN * LDS_S];

  int Mrows = g.dynM ? *g.dynM : g.M;
  int mBlocks = (Mrows + BM - 1) / BM;
  if ((int)blockIdx.y >= mBlocks) return;        // uniform exit

  int t = threadIdx.x;
  int wid = t >> 5, lane = t & 31;
  int wm = wid >> 1, wn = wid & 1;
  int half = (lane >> 4) & 1, l16 = lane & 15;
  int m0 = blockIdx.y * BM;
  int n0 = blockIdx.x * BN;

  // global->LDS staging map: 2 passes of 16B per thread per matrix
  const unsigned short* aPtr[2]; bool aVal[2]; int aSt[2];
  const unsigned short* bPtr[2]; int bSt[2];
  #pragma unroll
  for (int p = 0; p < 2; ++p) {
    int idx = t + p * 256;
    int r = idx >> 2;            // 0..127
    int kc = (idx & 3) * 8;      // 0,8,16,24 (bf16 elems)
    int gr = m0 + r;
    bool valid = gr < Mrows;
    int arow = valid ? (g.rowIdx ? g.rowIdx[gr] : gr) : 0;
    aPtr[p] = g.A + (size_t)arow * g.lda + kc;
    aVal[p] = valid;
    aSt[p] = r * LDS_S + kc;
    bPtr[p] = g.Bw + (size_t)(n0 + r) * g.ldb + kc;
    bSt[p] = r * LDS_S + kc;
  }
  (void)bPtr; (void)bSt;
  const unsigned short* bTile = g.Bw + (size_t)n0 * g.ldb;

  int nK = g.K / BK;
#if HAVE_ASYNC_LDS
  // prologue: stage tile 0
  #pragma unroll
  for (int p = 0; p < 2; ++p)
    if (aVal[p]) async_cp16(aPtr[p], &As[0][aSt[p]]);
#if USE_TDM
  if (wid == 0) tdm_load_B(bTile, &Bs[0][0], g.K, g.N, g.ldb);
#else
  #pragma unroll
  for (int p = 0; p < 2; ++p) async_cp16(bPtr[p], &Bs[0][bSt[p]]);
#endif
#else
  uint4 zr; zr.x = zr.y = zr.z = zr.w = 0u;
  uint4 ar[2], br[2];
  #pragma unroll
  for (int p = 0; p < 2; ++p) {
    ar[p] = aVal[p] ? *(const uint4*)(aPtr[p]) : zr;
    br[p] = *(const uint4*)(bPtr[p]);
    *(uint4*)&As[0][aSt[p]] = ar[p];
    *(uint4*)&Bs[0][bSt[p]] = br[p];
  }
#endif

  v8f acc[2][4] = {};

  for (int kt = 0; kt < nK; ++kt) {
#if HAVE_ASYNC_LDS
    wait_async0();               // my tile-kt async loads are in LDS
#endif
#if USE_TDM
    wait_tensor0();              // wave0's B-tile DMA done (no-op elsewhere)
#endif
    __syncthreads();             // everyone's are
    int cur = kt & 1, nxt = cur ^ 1;
    if (kt + 1 < nK) {
      size_t off = (size_t)(kt + 1) * BK;
#if HAVE_ASYNC_LDS
      #pragma unroll
      for (int p = 0; p < 2; ++p)
        if (aVal[p]) async_cp16(aPtr[p] + off, &As[nxt][aSt[p]]);
#if USE_TDM
      if (wid == 0) tdm_load_B(bTile + off, &Bs[nxt][0], g.K, g.N, g.ldb);
#else
      #pragma unroll
      for (int p = 0; p < 2; ++p) async_cp16(bPtr[p] + off, &Bs[nxt][bSt[p]]);
#endif
#else
      #pragma unroll
      for (int p = 0; p < 2; ++p) {
        ar[p] = aVal[p] ? *(const uint4*)(aPtr[p] + off) : zr;
        br[p] = *(const uint4*)(bPtr[p] + off);
      }
#endif
      if (kt + 2 < nK) {
        __builtin_prefetch(aPtr[0] + (size_t)(kt + 2) * BK, 0, 1);
        __builtin_prefetch(bTile + (size_t)(kt + 2) * BK, 0, 1);
      }
    }

    FragAB a[2], b[4];
    #pragma unroll
    for (int mf = 0; mf < 2; ++mf) {
      int m = wm * 32 + mf * 16 + l16;
      int klo = half * 8;
      a[mf].u[0] = *(const uint4*)&As[cur][m * LDS_S + klo];
      a[mf].u[1] = *(const uint4*)&As[cur][m * LDS_S + klo + 16];
    }
    #pragma unroll
    for (int nf = 0; nf < 4; ++nf) {
      int n = wn * 64 + nf * 16 + l16;
      int ks = half * 8;
      b[nf].u[0] = *(const uint4*)&Bs[cur][n * LDS_S + ks];
      b[nf].u[1] = *(const uint4*)&Bs[cur][n * LDS_S + ks + 16];
    }
    #pragma unroll
    for (int mf = 0; mf < 2; ++mf)
      #pragma unroll
      for (int nf = 0; nf < 4; ++nf)
        acc[mf][nf] = __builtin_amdgcn_wmma_f32_16x16x32_bf16(
            false, a[mf].v, false, b[nf].v, (short)0, acc[mf][nf], false, false);

#if !HAVE_ASYNC_LDS
    if (kt + 1 < nK) {
      #pragma unroll
      for (int p = 0; p < 2; ++p) {
        *(uint4*)&As[nxt][aSt[p]] = ar[p];
        *(uint4*)&Bs[nxt][bSt[p]] = br[p];
      }
    }
#endif
  }

  // epilogue
  #pragma unroll
  for (int mf = 0; mf < 2; ++mf) {
    #pragma unroll
    for (int nf = 0; nf < 4; ++nf) {
      #pragma unroll
      for (int r = 0; r < 8; ++r) {
        int m = m0 + wm * 32 + mf * 16 + r + half * 8;
        int n = n0 + wn * 64 + nf * 16 + l16;
        if (m >= Mrows) continue;
        float vv = acc[mf][nf][r];
        if (EPI == 0) {               // qkv split -> bf16 buffers
          int part = n >> 11, col = n & 2047;
          unsigned short* dst = (part == 0) ? g.q : (part == 1 ? g.kk : g.v);
          dst[(size_t)m * 2048 + col] = f2bf(vv);
        } else if (EPI == 1) {        // + residual -> f32 x1 and d_out
          float rr = vv + g.resid[(size_t)m * g.N + n];
          g.x1[(size_t)m * g.N + n] = rr;
          g.out2[(size_t)m * g.N + n] = rr;
        } else if (EPI == 2) {        // silu -> bf16 hx
          float s = vv / (1.f + __expf(-vv));
          g.hx[(size_t)m * g.N + n] = f2bf(s);
        } else {                      // EPI 3: weighted atomic scatter
          float w = g.slotW[m];
          int tok = g.slotTok[m];
          atomicAdd(g.atom + (size_t)tok * 2048 + n, w * vv);
        }
      }
    }
  }
}

// ---------------- flash attention (bf16 WMMA, D=128, S=1024, causal) ----------------
#define KS_S 136
#define VT_S 40
__global__ __launch_bounds__(256)
void k_attn(const unsigned short* __restrict__ qb,
            const unsigned short* __restrict__ kb,
            const unsigned short* __restrict__ vb,
            const int* __restrict__ amask,
            unsigned short* __restrict__ ob) {
  __shared__ unsigned short Ks[32 * KS_S];
  __shared__ unsigned short Vt[128 * VT_S];
  __shared__ unsigned short Pt[8 * 16 * 40];

  int t = threadIdx.x, wid = t >> 5, lane = t & 31;
  int half = (lane >> 4) & 1, l16 = lane & 15;
  int qblk = blockIdx.x, h = blockIdx.y, b = blockIdx.z;
  size_t base = ((size_t)(b * 16 + h)) * 1024 * 128;
  const int* am = amask + b * 1024;

  // Q fragments held in registers (A-operand layout)
  FragAB qf[4];
  int qrow = qblk * 128 + wid * 16 + l16;
  const unsigned short* qp = qb + base + (size_t)qrow * 128;
  #pragma unroll
  for (int c = 0; c < 4; ++c) {
    int klo = c * 32 + half * 8;
    qf[c].u[0] = *(const uint4*)(qp + klo);
    qf[c].u[1] = *(const uint4*)(qp + klo + 16);
  }

  v8f o[8] = {};
  float mx[8], ls[8];
  #pragma unroll
  for (int r = 0; r < 8; ++r) { mx[r] = -1e30f; ls[r] = 0.f; }

  const float rs = 0.08838834764831845f;   // 1/sqrt(128)
  int nkv = (qblk + 1) * 4;
  for (int j = 0; j < nkv; ++j) {
    int key0 = j * 32;
    {   // K tile: async direct-to-LDS; V tile: load + transposed LDS store
      int kr = t >> 3, dc = (t & 7) * 16;
      const unsigned short* kp = kb + base + (size_t)(key0 + kr) * 128 + dc;
#if HAVE_ASYNC_LDS
      async_cp16(kp,     &Ks[kr * KS_S + dc]);
      async_cp16(kp + 8, &Ks[kr * KS_S + dc + 8]);
#else
      *(uint4*)&Ks[kr * KS_S + dc]     = *(const uint4*)kp;
      *(uint4*)&Ks[kr * KS_S + dc + 8] = *(const uint4*)(kp + 8);
#endif
      const unsigned short* vp = vb + base + (size_t)(key0 + kr) * 128 + dc;
      uint4 v0 = *(const uint4*)vp, v1 = *(const uint4*)(vp + 8);
      const unsigned short* e0 = (const unsigned short*)&v0;
      const unsigned short* e1 = (const unsigned short*)&v1;
      #pragma unroll
      for (int i = 0; i < 8; ++i) Vt[(dc + i) * VT_S + kr] = e0[i];
      #pragma unroll
      for (int i = 0; i < 8; ++i) Vt[(dc + 8 + i) * VT_S + kr] = e1[i];
    }
#if HAVE_ASYNC_LDS
    wait_async0();
#endif
    __syncthreads();

    // S = Q * K^T   (16 q-rows x 32 keys per wave)
    v8f s[2] = {};
    #pragma unroll
    for (int c = 0; c < 4; ++c) {
      #pragma unroll
      for (int fn = 0; fn < 2; ++fn) {
        FragAB bk;
        int n = fn * 16 + l16;
        int ks = c * 32 + half * 8;
        bk.u[0] = *(const uint4*)&Ks[n * KS_S + ks];
        bk.u[1] = *(const uint4*)&Ks[n * KS_S + ks + 16];
        s[fn] = __builtin_amdgcn_wmma_f32_16x16x32_bf16(
            false, qf[c].v, false, bk.v, (short)0, s[fn], false, false);
      }
    }

    // mask + online softmax
    float pv[2][8];
    #pragma unroll
    for (int fn = 0; fn < 2; ++fn) {
      int key = key0 + fn * 16 + l16;
      int msk = am[key];
      #pragma unroll
      for (int r = 0; r < 8; ++r) {
        int qr = qblk * 128 + wid * 16 + r + half * 8;
        float sc = s[fn][r] * rs;
        sc = (msk == 0) ? NEGC : sc;
        sc += (key <= qr) ? 1.0f : NEGC;
        pv[fn][r] = sc;
      }
    }
    #pragma unroll
    for (int r = 0; r < 8; ++r) {
      float m2 = fmaxf(pv[0][r], pv[1][r]);
      #pragma unroll
      for (int d = 8; d; d >>= 1) m2 = fmaxf(m2, __shfl_xor(m2, d, 32));
      float mn = fmaxf(mx[r], m2);
      float scale = __expf(mx[r] - mn);
      pv[0][r] = __expf(pv[0][r] - mn);
      pv[1][r] = __expf(pv[1][r] - mn);
      float ps = pv[0][r] + pv[1][r];
      #pragma unroll
      for (int d = 8; d; d >>= 1) ps += __shfl_xor(ps, d, 32);
      ls[r] = ls[r] * scale + ps;
      mx[r] = mn;
      #pragma unroll
      for (int nf = 0; nf < 8; ++nf) o[nf][r] *= scale;
    }

    // P -> LDS (A-operand layout), then O += P * V
    unsigned short* Pw = Pt + wid * 16 * 40;
    #pragma unroll
    for (int fn = 0; fn < 2; ++fn)
      #pragma unroll
      for (int r = 0; r < 8; ++r)
        Pw[(r + half * 8) * 40 + fn * 16 + l16] = f2bf(pv[fn][r]);

    FragAB pa;
    pa.u[0] = *(const uint4*)&Pw[l16 * 40 + half * 8];
    pa.u[1] = *(const uint4*)&Pw[l16 * 40 + half * 8 + 16];
    #pragma unroll
    for (int nf = 0; nf < 8; ++nf) {
      FragAB vf;
      int d = nf * 16 + l16;
      vf.u[0] = *(const uint4*)&Vt[d * VT_S + half * 8];
      vf.u[1] = *(const uint4*)&Vt[d * VT_S + half * 8 + 16];
      o[nf] = __builtin_amdgcn_wmma_f32_16x16x32_bf16(
          false, pa.v, false, vf.v, (short)0, o[nf], false, false);
    }
    __syncthreads();
  }

  // normalize + store O (bf16, (B,H,S,D) flat == token rows)
  #pragma unroll
  for (int nf = 0; nf < 8; ++nf) {
    #pragma unroll
    for (int r = 0; r < 8; ++r) {
      int qr = qblk * 128 + wid * 16 + r + half * 8;
      int d = nf * 16 + l16;
      ob[base + (size_t)qr * 128 + d] = f2bf(o[nf][r] / ls[r]);
    }
  }
}

// ---------------- gating: softmax(xf @ gate_W^T), top-2, slot assignment ----------------
__global__ __launch_bounds__(256)
void k_gate(const unsigned short* __restrict__ h2,
            const float* __restrict__ gw,
            float* __restrict__ probsSum,
            int* __restrict__ expCount,
            int* __restrict__ rowIdxA,     // [8][4096]
            float* __restrict__ slotWA) {  // [8][4096]
  __shared__ float lg[8];
  int tok = blockIdx.x, t = threadIdx.x, w = t >> 5, lane = t & 31;
  const unsigned short* xr = h2 + (size_t)tok * 2048;
  const float* g = gw + (size_t)w * 2048;
  float acc = 0.f;
  for (int i = lane; i < 2048; i += 32) acc += bf2f(xr[i]) * g[i];
  #pragma unroll
  for (int m = 16; m; m >>= 1) acc += __shfl_xor(acc, m, 32);
  if (lane == 0) lg[w] = acc;
  __syncthreads();
  if (t == 0) {
    float m = lg[0];
    #pragma unroll
    for (int e = 1; e < 8; ++e) m = fmaxf(m, lg[e]);
    float p[8], s = 0.f;
    #pragma unroll
    for (int e = 0; e < 8; ++e) { p[e] = __expf(lg[e] - m); s += p[e]; }
    float inv = 1.f / s;
    int i0 = 0; float b0 = -1.f;
    #pragma unroll
    for (int e = 0; e < 8; ++e) {
      p[e] *= inv;
      atomicAdd(&probsSum[e], p[e]);
      if (p[e] > b0) { b0 = p[e]; i0 = e; }
    }
    int i1 = 0; float b1 = -1.f;
    #pragma unroll
    for (int e = 0; e < 8; ++e)
      if (e != i0 && p[e] > b1) { b1 = p[e]; i1 = e; }
    int s0 = atomicAdd(&expCount[i0], 1);
    rowIdxA[i0 * 4096 + s0] = tok; slotWA[i0 * 4096 + s0] = b0;
    int s1 = atomicAdd(&expCount[i1], 1);
    rowIdxA[i1 * 4096 + s1] = tok; slotWA[i1 * 4096 + s1] = b1;
  }
}

__global__ void k_zero8(float* ps, int* ec) {
  int t = threadIdx.x;
  if (t < 8) { ps[t] = 0.f; ec[t] = 0; }
}

__global__ void k_loss(const float* __restrict__ probsSum, float* __restrict__ out) {
  if (threadIdx.x == 0) {
    float p[8], mean = 0.f;
    #pragma unroll
    for (int e = 0; e < 8; ++e) { p[e] = probsSum[e] * (1.f / 4096.f); mean += p[e]; }
    mean *= 0.125f;
    float var = 0.f;
    #pragma unroll
    for (int e = 0; e < 8; ++e) { float d = p[e] - mean; var += d * d; }
    var *= 0.125f;
    out[0] = var / (mean * mean + 1e-10f);
  }
}

// ---------------- host ----------------
extern "C" void kernel_launch(void* const* d_in, const int* in_sizes, int n_in,
                              void* d_out, int out_size, void* d_ws, size_t ws_size,
                              hipStream_t stream) {
  (void)in_sizes; (void)n_in; (void)out_size; (void)ws_size;
  const float* x     = (const float*)d_in[0];
  const int*   amask = (const int*)d_in[1];
  const float* n1w   = (const float*)d_in[2];
  const float* qkvW  = (const float*)d_in[3];
  const float* outW  = (const float*)d_in[4];
  const float* n2w   = (const float*)d_in[5];
  const float* gateW = (const float*)d_in[6];
  const float* W1    = (const float*)d_in[7];
  const float* W2    = (const float*)d_in[8];
  float* out = (float*)d_out;

  char* ws = (char*)d_ws;
  size_t off = 0;
  auto alloc = [&](size_t bytes) -> char* {
    char* p = ws + off;
    off += (bytes + 255) & ~(size_t)255;
    return p;
  };
  const size_t NT = 4096;       // tokens
  unsigned short* qkvWb = (unsigned short*)alloc(6144ull * 2048 * 2);
  unsigned short* outWb = (unsigned short*)alloc(2048ull * 2048 * 2);
  unsigned short* W1b   = (unsigned short*)alloc(8ull * 1024 * 2048 * 2);
  unsigned short* W2b   = (unsigned short*)alloc(8ull * 2048 * 1024 * 2);
  unsigned short* hb    = (unsigned short*)alloc(NT * 2048 * 2);       // h, reused as h2
  char* big             = alloc(NT * 2048 * 2 * 4);                    // q,k,v,o -> reused as hx
  unsigned short* qb  = (unsigned short*)big;
  unsigned short* kbf = qb + NT * 2048;
  unsigned short* vbf = kbf + NT * 2048;
  unsigned short* obf = vbf + NT * 2048;
  unsigned short* hx  = (unsigned short*)big;                          // alias (q/k/v/o dead)
  float* x1       = (float*)alloc(NT * 2048 * 4);
  int*   rowIdxA  = (int*)alloc(8 * 4096 * 4);
  float* slotWA   = (float*)alloc(8 * 4096 * 4);
  float* probsSum = (float*)alloc(64);
  int*   expCount = (int*)alloc(64);

  k_zero8<<<1, 32, 0, stream>>>(probsSum, expCount);

  // weight conversion f32 -> bf16
  k_cvt<<<(6144 * 2048 / 4 + 255) / 256, 256, 0, stream>>>(qkvW, qkvWb, 6144 * 2048 / 4);
  k_cvt<<<(2048 * 2048 / 4 + 255) / 256, 256, 0, stream>>>(outW, outWb, 2048 * 2048 / 4);
  k_cvt<<<(8 * 1024 * 2048 / 4 + 255) / 256, 256, 0, stream>>>(W1, W1b, 8 * 1024 * 2048 / 4);
  k_cvt<<<(8 * 2048 * 1024 / 4 + 255) / 256, 256, 0, stream>>>(W2, W2b, 8 * 2048 * 1024 / 4);

  // h = rmsnorm(x) -> bf16
  k_rmsnorm<<<4096, 256, 0, stream>>>(x, n1w, hb);

  // qkv = h @ qkv_W^T  (split into q/k/v bf16)
  {
    GemmArgs a{};
    a.A = hb; a.lda = 2048; a.Bw = qkvWb; a.ldb = 2048;
    a.M = 4096; a.N = 6144; a.K = 2048;
    a.q = qb; a.kk = kbf; a.v = vbf;
    k_gemm<0><<<dim3(48, 32), 256, 0, stream>>>(a);
  }

  // attention
  k_attn<<<dim3(8, 16, 4), 256, 0, stream>>>(qb, kbf, vbf, amask, obf);

  // x1 = o @ out_W^T + x   (also primes d_out with res2)
  {
    GemmArgs a{};
    a.A = obf; a.lda = 2048; a.Bw = outWb; a.ldb = 2048;
    a.M = 4096; a.N = 2048; a.K = 2048;
    a.x1 = x1; a.out2 = out; a.resid = x;
    k_gemm<1><<<dim3(16, 32), 256, 0, stream>>>(a);
  }

  // h2 = rmsnorm(x1) -> bf16 (reuse hb)
  k_rmsnorm<<<4096, 256, 0, stream>>>(x1, n2w, hb);

  // gating + top-2 slot assignment + probs accumulation
  k_gate<<<4096, 256, 0, stream>>>(hb, gateW, probsSum, expCount, rowIdxA, slotWA);

  // expert FFN, gathered rows per expert
  for (int e = 0; e < 8; ++e) {
    GemmArgs a{};
    a.A = hb; a.lda = 2048;
    a.Bw = W1b + (size_t)e * 1024 * 2048; a.ldb = 2048;
    a.M = 4096; a.N = 1024; a.K = 2048;
    a.dynM = expCount + e;
    a.rowIdx = rowIdxA + e * 4096;
    a.hx = hx + (size_t)e * 4096 * 1024;
    k_gemm<2><<<dim3(8, 32), 256, 0, stream>>>(a);
  }
  for (int e = 0; e < 8; ++e) {
    GemmArgs a{};
    a.A = hx + (size_t)e * 4096 * 1024; a.lda = 1024;
    a.Bw = W2b + (size_t)e * 2048 * 1024; a.ldb = 1024;
    a.M = 4096; a.N = 2048; a.K = 1024;
    a.dynM = expCount + e;
    a.slotW = slotWA + e * 4096;
    a.slotTok = rowIdxA + e * 4096;
    a.atom = out;
    k_gemm<3><<<dim3(16, 32), 256, 0, stream>>>(a);
  }

  // aux load-balance loss -> d_out[B*S*E]
  k_loss<<<1, 32, 0, stream>>>(probsSum, out + NT * 2048);
}
